// MultiHeadAttention_11175504904406
// MI455X (gfx1250) — matrix operand order repaired and checked by reading, not compiled, not run on hardware
//
#include <hip/hip_runtime.h>

// ---------------- types ----------------
typedef __bf16 bf16_t;
typedef __attribute__((ext_vector_type(16))) __bf16 v16bf;
typedef __attribute__((ext_vector_type(8)))  __bf16 v8bf;
typedef __attribute__((ext_vector_type(8)))  float  v8f;

// ---------------- problem constants ----------------
constexpr int DMODEL = 1024;
constexpr int SEQ    = 2048;
constexpr int NBATCH = 2;
constexpr int NHEADS = 16;
constexpr int HDIM   = 64;              // DMODEL / NHEADS
constexpr int MTOT   = NBATCH * SEQ;    // 4096 rows through the projections

// ---------------- helpers ----------------
static __device__ __forceinline__ bf16_t f2bf(float f) {
  // round-to-nearest-even fp32 -> bf16 using integer ops only
  unsigned u = __builtin_bit_cast(unsigned, f);
  u += 0x7fffu + ((u >> 16) & 1u);
  unsigned short hs = (unsigned short)(u >> 16);
  return __builtin_bit_cast(bf16_t, hs);
}

static __device__ __forceinline__ v16bf cat16(v8bf lo, v8bf hi) {
  return __builtin_shufflevector(lo, hi, 0, 1, 2, 3, 4, 5, 6, 7,
                                 8, 9, 10, 11, 12, 13, 14, 15);
}

// A-matrix fragment (16x32 bf16, MxK), per CDNA5 ISA layout:
//   lanes 0-15 : row M = lane,    K chunks {kh..kh+7, 16+kh..} with kh=0
//   lanes 16-31: row M = lane-16, same with kh=8
static __device__ __forceinline__ v16bf load_a_frag(const bf16_t* base, int ld,
                                                    int m0, int k0, int lane) {
  int row = m0 + (lane & 15);
  int kh  = (lane >> 4) * 8;
  const bf16_t* p = base + (size_t)row * ld + k0 + kh;
  return cat16(*(const v8bf*)(p), *(const v8bf*)(p + 16));
}

// B-matrix fragment (32x16 bf16, KxN) from K-contiguous ("NT") storage:
//   lanes 0-15 : col N = n0+lane,    K = k0+{0..7, 16..23}
//   lanes 16-31: col N = n0+lane-16, K = k0+{8..15, 24..31}
static __device__ __forceinline__ v16bf load_b_frag(const bf16_t* base, int ld,
                                                    int n0, int k0, int lane) {
  int col = n0 + (lane & 15);
  int ks  = k0 + (lane >> 4) * 8;
  const bf16_t* p = base + (size_t)col * ld + ks;
  return cat16(*(const v8bf*)(p), *(const v8bf*)(p + 16));
}

static __device__ __forceinline__ v8f wmma_bf16(v16bf a, v16bf b, v8f c) {
  return __builtin_amdgcn_wmma_f32_16x16x32_bf16(
      /*neg_a=*/false, a, /*neg_b=*/false, b,
      /*c_mod=*/(short)0, c, /*reuse_a=*/false, /*reuse_b=*/false);
}

// ---------------- fp32 -> bf16 convert ----------------
__global__ void cvt_f32_bf16(const float* __restrict__ in,
                             bf16_t* __restrict__ out, int n) {
  int i = blockIdx.x * blockDim.x + threadIdx.x;
  if (i < n) out[i] = f2bf(in[i]);
}

// ---------------- NT GEMM: C[m,n] = sum_k A[m,k]*B[n,k] + bias[n] ----------------
// 256 threads = 8 waves arranged 2(M) x 4(N); per-wave tile 64x64 -> block 128x256.
template <bool STORE_BF16>
__global__ __launch_bounds__(256)
__attribute__((amdgpu_waves_per_eu(2)))
void gemm_nt(const bf16_t* __restrict__ A, const bf16_t* __restrict__ B,
             const float* __restrict__ bias,
             float* __restrict__ Cf, bf16_t* __restrict__ Cb,
             int M, int N, int K) {
  const int lane = threadIdx.x & 31;
  const int wave = threadIdx.x >> 5;
  const int m0 = blockIdx.y * 128 + (wave >> 2) * 64;
  const int n0 = blockIdx.x * 256 + (wave & 3) * 64;

  v8f acc[4][4];
#pragma unroll
  for (int i = 0; i < 4; ++i)
#pragma unroll
    for (int j = 0; j < 4; ++j) acc[i][j] = (v8f){};

  for (int k0 = 0; k0 < K; k0 += 32) {
    if (k0 + 32 < K)  // hint next A panel (global_prefetch_b8)
      __builtin_prefetch(A + (size_t)(m0 + (lane & 15)) * K + k0 + 32, 0, 1);

    v16bf bfr[4], af[4];
#pragma unroll
    for (int j = 0; j < 4; ++j) bfr[j] = load_b_frag(B, K, n0 + 16 * j, k0, lane);
#pragma unroll
    for (int i = 0; i < 4; ++i) af[i] = load_a_frag(A, K, m0 + 16 * i, k0, lane);
#pragma unroll
    for (int i = 0; i < 4; ++i)
#pragma unroll
      for (int j = 0; j < 4; ++j) acc[i][j] = wmma_bf16(af[i], bfr[j], acc[i][j]);
  }

  // epilogue: C/D layout -> VGPR v holds rows {v, v+8} split by lane halves
  const int colb   = lane & 15;
  const int rowoff = (lane >> 4) * 8;
#pragma unroll
  for (int j = 0; j < 4; ++j) {
    int col = n0 + j * 16 + colb;
    float bv = bias[col];
#pragma unroll
    for (int i = 0; i < 4; ++i) {
#pragma unroll
      for (int v = 0; v < 8; ++v) {
        int row = m0 + i * 16 + rowoff + v;
        float val = acc[i][j][v] + bv;
        if (STORE_BF16) Cb[(size_t)row * N + col] = f2bf(val);
        else            Cf[(size_t)row * N + col] = val;
      }
    }
  }
}

// ---------------- flash attention over proj (Q=K=V=proj heads) ----------------
// grid: (SEQ/128, NHEADS, NBATCH); block: 256 threads = 8 waves, 16 queries each.
// The 64-key tile (K and V are the same rows of proj) is loaded from global ONCE
// per block and stored to LDS twice: row-major KT (QK^T B-frags) and transposed
// VT (PV B-frags). All WMMA operands then come from short-latency ds_load_b128.
__global__ __launch_bounds__(256)
__attribute__((amdgpu_waves_per_eu(2)))
void attn_kernel(const bf16_t* __restrict__ proj, bf16_t* __restrict__ vals) {
  const int lane = threadIdx.x & 31;
  const int wave = threadIdx.x >> 5;
  const int h = blockIdx.y;
  const int b = blockIdx.z;
  const bf16_t* ph = proj + (size_t)b * SEQ * DMODEL + h * HDIM;  // row stride DMODEL

  __shared__ __align__(16) bf16_t KT[64 * HDIM];    // K tile: [key][d]   64x64
  __shared__ __align__(16) bf16_t VT[HDIM * 64];    // V^T tile: [d][key] 64x64
  __shared__ __align__(16) bf16_t PS[8][16 * 64];   // per-wave P tile: [qrow][key]

  const int q0 = blockIdx.x * 128 + wave * 16;

  // Q fragments for this wave's 16 query rows (Dh=64 -> two K-chunks of 32)
  v16bf qf0 = load_a_frag(ph, DMODEL, q0, 0, lane);
  v16bf qf1 = load_a_frag(ph, DMODEL, q0, 32, lane);

  float m[8], l[8];
  v8f O[4];
#pragma unroll
  for (int j = 0; j < 8; ++j) { m[j] = -1e30f; l[j] = 0.f; }
#pragma unroll
  for (int n = 0; n < 4; ++n) O[n] = (v8f){};

  // staging coordinates: 256 threads cover 64 keys x 64 d, 16 bf16 per thread
  const int skey = threadIdx.x >> 2;        // 0..63
  const int sd   = (threadIdx.x & 3) * 16;  // 0,16,32,48

  for (int k0 = 0; k0 < SEQ; k0 += 64) {
    __syncthreads();  // protect KT/VT against previous-iteration readers
    {
      const bf16_t* src = ph + (size_t)(k0 + skey) * DMODEL + sd;
      v8bf e0 = *(const v8bf*)(src);
      v8bf e1 = *(const v8bf*)(src + 8);
      *(v8bf*)&KT[skey * 64 + sd]     = e0;          // row-major for QK^T
      *(v8bf*)&KT[skey * 64 + sd + 8] = e1;
#pragma unroll
      for (int i = 0; i < 8; ++i) {                  // transposed for PV
        VT[(sd + i) * 64 + skey]     = e0[i];
        VT[(sd + 8 + i) * 64 + skey] = e1[i];
      }
      if (k0 + 64 < SEQ)  // prefetch next key tile (global_prefetch_b8)
        __builtin_prefetch(ph + (size_t)(k0 + 64 + skey) * DMODEL + sd, 0, 1);
    }
    __syncthreads();

    // ---- preload all K fragments from LDS (16 ds_load_b128 in flight) ----
    v16bf kf[4][2];
#pragma unroll
    for (int t = 0; t < 4; ++t) {
      kf[t][0] = load_b_frag(KT, 64, t * 16, 0, lane);
      kf[t][1] = load_b_frag(KT, 64, t * 16, 32, lane);
    }

    // ---- scores S = (Q K^T) * Dh^-0.5 for 4 key sub-tiles of 16 ----
    v8f sc[4];
#pragma unroll
    for (int t = 0; t < 4; ++t) {
      v8f s = (v8f){};
      s = wmma_bf16(qf0, kf[t][0], s);
      s = wmma_bf16(qf1, kf[t][1], s);
#pragma unroll
      for (int j = 0; j < 8; ++j) s[j] *= 0.125f;  // 1/sqrt(64)
      sc[t] = s;
    }

    // ---- online softmax: row stats via 16-lane xor reductions ----
#pragma unroll
    for (int j = 0; j < 8; ++j) {
      float rm = fmaxf(fmaxf(sc[0][j], sc[1][j]), fmaxf(sc[2][j], sc[3][j]));
      rm = fmaxf(rm, __shfl_xor(rm, 1));
      rm = fmaxf(rm, __shfl_xor(rm, 2));
      rm = fmaxf(rm, __shfl_xor(rm, 4));
      rm = fmaxf(rm, __shfl_xor(rm, 8));
      float mn = fmaxf(m[j], rm);
      float alpha = __expf(m[j] - mn);
      m[j] = mn;
      float rs = 0.f;
#pragma unroll
      for (int t = 0; t < 4; ++t) {
        float p = __expf(sc[t][j] - mn);
        sc[t][j] = p;
        rs += p;
      }
      rs += __shfl_xor(rs, 1);
      rs += __shfl_xor(rs, 2);
      rs += __shfl_xor(rs, 4);
      rs += __shfl_xor(rs, 8);
      l[j] = l[j] * alpha + rs;
#pragma unroll
      for (int n = 0; n < 4; ++n) O[n][j] *= alpha;
    }

    // ---- P: C-layout -> LDS -> A-layout (LDS is in-order within a wave) ----
    bf16_t* pw = &PS[wave][0];
    {
      int prow = (lane >> 4) * 8;
      int pcol = lane & 15;
#pragma unroll
      for (int j = 0; j < 8; ++j)
#pragma unroll
        for (int t = 0; t < 4; ++t)
          pw[(prow + j) * 64 + t * 16 + pcol] = f2bf(sc[t][j]);
    }

    // ---- preload P (A-layout) and V^T (B-layout) fragments from LDS ----
    v16bf pf0 = load_a_frag(pw, 64, 0, 0, lane);   // keys 0..31 of this tile
    v16bf pf1 = load_a_frag(pw, 64, 0, 32, lane);  // keys 32..63
    v16bf vf[4][2];
#pragma unroll
    for (int n = 0; n < 4; ++n) {
      vf[n][0] = load_b_frag(VT, 64, n * 16, 0, lane);
      vf[n][1] = load_b_frag(VT, 64, n * 16, 32, lane);
    }

    // ---- O += P @ V ----
#pragma unroll
    for (int n = 0; n < 4; ++n) {
      O[n] = wmma_bf16(pf0, vf[n][0], O[n]);
      O[n] = wmma_bf16(pf1, vf[n][1], O[n]);
    }
  }

  // ---- epilogue: O / l -> vals (bf16, [b, q, h*64+d]) ----
  {
    int rowoff = (lane >> 4) * 8;
    int col = lane & 15;
#pragma unroll
    for (int j = 0; j < 8; ++j) {
      float inv = 1.0f / l[j];
      int row = q0 + rowoff + j;
      size_t rbase = ((size_t)b * SEQ + row) * DMODEL + h * HDIM;
#pragma unroll
      for (int n = 0; n < 4; ++n)
        vals[rbase + n * 16 + col] = f2bf(O[n][j] * inv);
    }
  }
}

// ---------------- launch ----------------
extern "C" void kernel_launch(void* const* d_in, const int* in_sizes, int n_in,
                              void* d_out, int out_size, void* d_ws, size_t ws_size,
                              hipStream_t stream) {
  const float* q  = (const float*)d_in[0];
  const float* Wq = (const float*)d_in[1];
  const float* bq = (const float*)d_in[2];
  const float* Wo = (const float*)d_in[3];
  const float* bo = (const float*)d_in[4];
  float* out = (float*)d_out;

  // workspace layout (bf16): q | Wq | Wo | proj | vals  (~29.4 MB)
  bf16_t* q_bf  = (bf16_t*)d_ws;
  bf16_t* Wq_bf = q_bf  + (size_t)MTOT * DMODEL;
  bf16_t* Wo_bf = Wq_bf + (size_t)DMODEL * DMODEL;
  bf16_t* projb = Wo_bf + (size_t)DMODEL * DMODEL;
  bf16_t* valsb = projb + (size_t)MTOT * DMODEL;

  const int nq = MTOT * DMODEL;    // 4.19M
  const int nw = DMODEL * DMODEL;  // 1.05M
  cvt_f32_bf16<<<(nq + 255) / 256, 256, 0, stream>>>(q, q_bf, nq);
  cvt_f32_bf16<<<(nw + 255) / 256, 256, 0, stream>>>(Wq, Wq_bf, nw);
  cvt_f32_bf16<<<(nw + 255) / 256, 256, 0, stream>>>(Wo, Wo_bf, nw);

  // proj = bf16(q @ Wq^T + bq)
  gemm_nt<true><<<dim3(DMODEL / 256, MTOT / 128), 256, 0, stream>>>(
      q_bf, Wq_bf, bq, nullptr, projb, MTOT, DMODEL, DMODEL);

  // attention (Q=K=V=proj heads), output bf16
  attn_kernel<<<dim3(SEQ / 128, NHEADS, NBATCH), 256, 0, stream>>>(projb, valsb);

  // out = vals @ Wo^T + bo (fp32)
  gemm_nt<false><<<dim3(DMODEL / 256, MTOT / 128), 256, 0, stream>>>(
      valsb, Wo_bf, bo, out, nullptr, MTOT, DMODEL, DMODEL);
}
